// Quant3Linear_79723182949081
// MI455X (gfx1250) — matrix operand order repaired and checked by reading, not compiled, not run on hardware
//
#include <hip/hip_runtime.h>
#include <hip/hip_bf16.h>
#include <stdint.h>

// ---------------------------------------------------------------------------
// Quant2 linear with input projection + permutations, CDNA5 (gfx1250) wave32.
//   W2[r,j] = sum_k deq(Wq[perm_out[r],k]) * P_in[perm_in[j],k]   (GEMM1)
//   out[m,o] = sum_i x[m,i] * W2[o,i] + bias[o]                   (GEMM2)
// NT GEMMs on v_wmma_f32_16x16x32_bf16, f32 accumulation, BK=64 stages.
// Workspace: W2 stored bf16, 4096*4096*2 = 32 MB.
// ---------------------------------------------------------------------------

#define IN_DIM   4096
#define OUT_DIM  4096
#define XROWS    8192      // 4 * 2048
#define BM       128
#define BN       128
#define BK       64
#define LDSS     72        // 64 + 8 pad (144B rows -> conflict-free b128 reads)

typedef __attribute__((ext_vector_type(16))) __bf16 v16bf;
typedef __attribute__((ext_vector_type(8)))  __bf16 v8bf;
typedef __attribute__((ext_vector_type(8)))  float  v8f;

__device__ __forceinline__ unsigned short bfbits(float f) {
    return __builtin_bit_cast(unsigned short, (__bf16)f);   // native v_cvt, RNE
}

// A fragment (16-bit A 16x32 layout): lane&15 = M row; k0 = (lane>>4)*8;
// VGPR0-3 <- K=k0..k0+7, VGPR4-7 <- K=k0+16..k0+23.
__device__ __forceinline__ v16bf load_frag_a(const unsigned short* lds, int rowBase,
                                             int kk, int lane) {
    int m  = rowBase + (lane & 15);
    int k0 = kk + (lane >> 4) * 8;
    const v8bf lo = *(const v8bf*)(lds + m * LDSS + k0);
    const v8bf hi = *(const v8bf*)(lds + m * LDSS + k0 + 16);
    v16bf r;
#pragma unroll
    for (int i = 0; i < 8; ++i) { r[i] = lo[i]; r[i + 8] = hi[i]; }
    return r;
}

// B fragment: lane&15 = N col; k0 = (lane>>4)*8;
// VGPR0-3 <- K=k0..k0+7, VGPR4-7 <- K=k0+16..k0+23 (same as A), N-major LDS tile.
__device__ __forceinline__ v16bf load_frag_b(const unsigned short* lds, int colBase,
                                             int kk, int lane) {
    int n  = colBase + (lane & 15);
    int k0 = kk + (lane >> 4) * 8;
    const v8bf lo = *(const v8bf*)(lds + n * LDSS + k0);
    const v8bf hi = *(const v8bf*)(lds + n * LDSS + k0 + 16);
    v16bf r;
#pragma unroll
    for (int i = 0; i < 8; ++i) { r[i] = lo[i]; r[i + 8] = hi[i]; }
    return r;
}

// ---------------------------------------------------------------------------
// GEMM1: fused dequant + gathered projection.  Grid (32, 32), 256 threads.
// ---------------------------------------------------------------------------
__global__ __launch_bounds__(256)
void wq_transform_gemm(const uint32_t* __restrict__ qweight,
                       const uint32_t* __restrict__ zeros_packed,
                       const float*    __restrict__ scale,
                       const float*    __restrict__ P_in,
                       const int*      __restrict__ perm_in,
                       const int*      __restrict__ perm_out,
                       unsigned short* __restrict__ W2) {
    __shared__ unsigned short ldsA[BM * LDSS];
    __shared__ unsigned short ldsB[BN * LDSS];

    const int tid  = threadIdx.x;
    const int lane = tid & 31;
    const int wave = tid >> 5;
    const int wm   = (wave & 3) * 32;   // wave M offset inside tile
    const int wn   = (wave >> 2) * 64;  // wave N offset inside tile

    // A-side loader: 32 weights (2 packed dwords) per thread per stage
    const int rowA = tid >> 1;          // 0..127
    const int kwA  = tid & 1;           // which 32-wide half of the 64-K stage
    const int pr   = perm_out[blockIdx.x * BM + rowA];
    const uint32_t zw = zeros_packed[pr >> 4];
    const float zp = (float)((zw >> ((pr & 15) * 2)) & 3u);
    const float sc = scale[pr];
    const uint32_t* qrow = qweight + (size_t)pr * (IN_DIM / 16);

    // B-side loader: 32 gathered floats per thread per stage
    const int pc = perm_in[blockIdx.y * BN + rowA];
    const float* prow = P_in + (size_t)pc * IN_DIM + kwA * 32;

    v8f acc[2][4];
#pragma unroll
    for (int i = 0; i < 2; ++i)
#pragma unroll
        for (int j = 0; j < 4; ++j) acc[i][j] = 0.0f;

    for (int k0 = 0; k0 < IN_DIM; k0 += BK) {
        // A tile: dequant 2 dwords -> 32 bf16
        const uint2 q2 = *(const uint2*)(qrow + (k0 >> 4) + kwA * 2);
        unsigned short* da = &ldsA[rowA * LDSS + kwA * 32];
#pragma unroll
        for (int j = 0; j < 16; ++j) {
            da[j]      = bfbits(((float)((q2.x >> (2 * j)) & 3u) - zp) * sc);
            da[j + 16] = bfbits(((float)((q2.y >> (2 * j)) & 3u) - zp) * sc);
        }
        // B tile: gathered P_in row, f32 -> bf16 (32 values)
        const float4* ps = (const float4*)(prow + k0);
        unsigned short* db = &ldsB[rowA * LDSS + kwA * 32];
#pragma unroll
        for (int v = 0; v < 8; ++v) {
            float4 f = ps[v];
            db[4 * v + 0] = bfbits(f.x);
            db[4 * v + 1] = bfbits(f.y);
            db[4 * v + 2] = bfbits(f.z);
            db[4 * v + 3] = bfbits(f.w);
        }
        // speculative prefetch of next stage's streams (global_prefetch_b8)
        __builtin_prefetch((const void*)(prow + k0 + BK), 0, 0);
        __builtin_prefetch((const void*)(qrow + ((k0 + BK) >> 4) + kwA * 2), 0, 0);
        __syncthreads();

#pragma unroll
        for (int kk = 0; kk < BK; kk += 32) {
            v16bf af[2], bf[4];
#pragma unroll
            for (int ti = 0; ti < 2; ++ti) af[ti] = load_frag_a(ldsA, wm + ti * 16, kk, lane);
#pragma unroll
            for (int tn = 0; tn < 4; ++tn) bf[tn] = load_frag_b(ldsB, wn + tn * 16, kk, lane);
#pragma unroll
            for (int ti = 0; ti < 2; ++ti)
#pragma unroll
                for (int tn = 0; tn < 4; ++tn)
                    acc[ti][tn] = __builtin_amdgcn_wmma_f32_16x16x32_bf16(
                        false, af[ti], false, bf[tn], (short)0, acc[ti][tn], false, false);
        }
        __syncthreads();
    }

    // Epilogue: C layout -> W2 bf16 (row-major OUT x IN)
    const int rl  = lane & 15;
    const int rhi = (lane >> 4) * 8;
#pragma unroll
    for (int ti = 0; ti < 2; ++ti) {
        const int rowBase = blockIdx.x * BM + wm + ti * 16 + rhi;
#pragma unroll
        for (int tn = 0; tn < 4; ++tn) {
            const int col = blockIdx.y * BN + wn + tn * 16 + rl;
#pragma unroll
            for (int v = 0; v < 8; ++v)
                W2[(size_t)(rowBase + v) * IN_DIM + col] = bfbits(acc[ti][tn][v]);
        }
    }
}

// ---------------------------------------------------------------------------
// GEMM2: out = x @ W2^T + bias.  Grid (64, 32), 256 threads.
// ---------------------------------------------------------------------------
__global__ __launch_bounds__(256)
void xw_gemm(const float* __restrict__ x,
             const unsigned short* __restrict__ W2,
             const float* __restrict__ bias,
             float* __restrict__ out) {
    __shared__ unsigned short ldsA[BM * LDSS];
    __shared__ unsigned short ldsB[BN * LDSS];

    const int tid  = threadIdx.x;
    const int lane = tid & 31;
    const int wave = tid >> 5;
    const int wm   = (wave & 3) * 32;
    const int wn   = (wave >> 2) * 64;

    const int rowA = tid >> 1;
    const int half = tid & 1;
    const float* xrow = x + (size_t)(blockIdx.x * BM + rowA) * IN_DIM + half * 32;
    const unsigned short* wrow = W2 + (size_t)(blockIdx.y * BN + rowA) * IN_DIM + half * 32;

    v8f acc[2][4];
#pragma unroll
    for (int i = 0; i < 2; ++i)
#pragma unroll
        for (int j = 0; j < 4; ++j) acc[i][j] = 0.0f;

    for (int k0 = 0; k0 < IN_DIM; k0 += BK) {
        // A tile: x f32 -> bf16 (32 values per thread)
        const float4* ps = (const float4*)(xrow + k0);
        unsigned short* da = &ldsA[rowA * LDSS + half * 32];
#pragma unroll
        for (int v = 0; v < 8; ++v) {
            float4 f = ps[v];
            da[4 * v + 0] = bfbits(f.x);
            da[4 * v + 1] = bfbits(f.y);
            da[4 * v + 2] = bfbits(f.z);
            da[4 * v + 3] = bfbits(f.w);
        }
        // B tile: W2 already bf16, stream 64B per thread
        const uint4* ws = (const uint4*)(wrow + k0);
        uint4* db = (uint4*)&ldsB[rowA * LDSS + half * 32];
#pragma unroll
        for (int v = 0; v < 4; ++v) db[v] = ws[v];

        __builtin_prefetch((const void*)(xrow + k0 + BK), 0, 0);
        __builtin_prefetch((const void*)(wrow + k0 + BK), 0, 0);
        __syncthreads();

#pragma unroll
        for (int kk = 0; kk < BK; kk += 32) {
            v16bf af[2], bf[4];
#pragma unroll
            for (int ti = 0; ti < 2; ++ti) af[ti] = load_frag_a(ldsA, wm + ti * 16, kk, lane);
#pragma unroll
            for (int tn = 0; tn < 4; ++tn) bf[tn] = load_frag_b(ldsB, wn + tn * 16, kk, lane);
#pragma unroll
            for (int ti = 0; ti < 2; ++ti)
#pragma unroll
                for (int tn = 0; tn < 4; ++tn)
                    acc[ti][tn] = __builtin_amdgcn_wmma_f32_16x16x32_bf16(
                        false, af[ti], false, bf[tn], (short)0, acc[ti][tn], false, false);
        }
        __syncthreads();
    }

    const int rl  = lane & 15;
    const int rhi = (lane >> 4) * 8;
#pragma unroll
    for (int ti = 0; ti < 2; ++ti) {
        const int rowBase = blockIdx.x * BM + wm + ti * 16 + rhi;
#pragma unroll
        for (int tn = 0; tn < 4; ++tn) {
            const int col = blockIdx.y * BN + wn + tn * 16 + rl;
            const float bv = bias[col];
#pragma unroll
            for (int v = 0; v < 8; ++v)
                out[(size_t)(rowBase + v) * OUT_DIM + col] = acc[ti][tn][v] + bv;
        }
    }
}

// ---------------------------------------------------------------------------
// Inputs (setup_inputs order):
//  0 x (f32, 4*2048*4096)   1 qweight (i32)   2 zeros_packed (i32)
//  3 scale (f32)            4 bias (f32)      5 P_in (f32)
//  6 perm_in (i32)          7 perm_out (i32)
// Output: f32, 4*2048*4096.  Workspace: >= 32 MB (W2 bf16).
// ---------------------------------------------------------------------------
extern "C" void kernel_launch(void* const* d_in, const int* in_sizes, int n_in,
                              void* d_out, int out_size, void* d_ws, size_t ws_size,
                              hipStream_t stream) {
    (void)in_sizes; (void)n_in; (void)out_size; (void)ws_size;
    const float*    x        = (const float*)d_in[0];
    const uint32_t* qweight  = (const uint32_t*)d_in[1];
    const uint32_t* zeros    = (const uint32_t*)d_in[2];
    const float*    scale    = (const float*)d_in[3];
    const float*    bias     = (const float*)d_in[4];
    const float*    P_in     = (const float*)d_in[5];
    const int*      perm_in  = (const int*)d_in[6];
    const int*      perm_out = (const int*)d_in[7];
    float*          out      = (float*)d_out;
    unsigned short* W2       = (unsigned short*)d_ws;   // 4096*4096 bf16 = 32 MB

    dim3 g1(OUT_DIM / BM, IN_DIM / BN);
    wq_transform_gemm<<<g1, 256, 0, stream>>>(qweight, zeros, scale, P_in,
                                              perm_in, perm_out, W2);

    dim3 g2(XROWS / BM, OUT_DIM / BN);
    xw_gemm<<<g2, 256, 0, stream>>>(x, W2, bias, out);
}
